// RoPEMultiheadAttention_67989332296148
// MI455X (gfx1250) — compile-verified, matrix operand order checked by reading, not run on hardware
//
#include <hip/hip_runtime.h>
#include <hip/hip_bf16.h>

// RoPE multi-head attention, MI455X (gfx1250, wave32, WMMA).
// All GEMMs via v_wmma_f32_16x16x32_bf16 (fp32 accum). Working set (~80MB)
// fits in the 192MB L2 -> compute bound -> bf16 matrix path everywhere.
// v3: double-buffered GEMM LDS (1 barrier / K-step, global prefetch overlapped
// with WMMA), step-major batched softmax max-reduction (8 bpermutes in flight),
// softmax row-sums via WMMA against an all-ones fragment.

typedef __bf16 bf16;
typedef __attribute__((ext_vector_type(16))) __bf16       v16bf;
typedef __attribute__((ext_vector_type(8)))  float        v8f;
typedef __attribute__((ext_vector_type(4)))  unsigned int u32x4;

// 16x32 (A) / 32x16 (B) bf16 fragment: 8 VGPRs per lane, loaded as two b128s.
// Lane L: m/n = L&15, half = L>>4. VGPRs0-3 = k in [8*half, 8*half+8),
// VGPRs4-7 = k in [16+8*half, 16+8*half+8) -> byte offsets half*16, 32+half*16.
union Frag { v16bf v; u32x4 u[2]; };

static __device__ __forceinline__ Frag load_frag(const bf16* rowptr, int half) {
    Frag f;
    const bf16* p = rowptr + half * 8;
    f.u[0] = *(const u32x4*)p;
    f.u[1] = *(const u32x4*)(p + 16);
    return f;
}

static __device__ __forceinline__ v8f wmma_bf16(const Frag& a, const Frag& b, v8f c) {
    return __builtin_amdgcn_wmma_f32_16x16x32_bf16(false, a.v, false, b.v,
                                                   (short)0, c, false, false);
}

// ---------------------------------------------------------------- converts --
__global__ __launch_bounds__(256) void convert_f32_bf16(const float* __restrict__ in,
                                                        bf16* __restrict__ out, int n) {
    int i = blockIdx.x * 256 + threadIdx.x;
    if (i < n) out[i] = (bf16)in[i];
}

// out[c][r] = (bf16) in[r][c]   (rows x cols fp32 -> cols x rows bf16)
__global__ __launch_bounds__(256) void transpose_to_bf16(const float* __restrict__ in,
                                                         bf16* __restrict__ out,
                                                         int rows, int cols) {
    __shared__ float tile[32][33];
    int c0 = blockIdx.x * 32, r0 = blockIdx.y * 32;
    int tx = threadIdx.x, ty = threadIdx.y;           // block (32,8)
    for (int j = ty; j < 32; j += 8)
        tile[j][tx] = in[(size_t)(r0 + j) * cols + c0 + tx];
    __syncthreads();
    for (int j = ty; j < 32; j += 8)
        out[(size_t)(c0 + j) * rows + r0 + tx] = (bf16)tile[tx][j];
}

// ----------------------------------------------------------------- GEMM -----
// C[M][Nd] = A[M][Kd] * Bt[Nd][Kd]^T. 128x128 tile per WG, 8 waves, each wave
// a 32x64 slab: per 32-wide K-step 2 A-frags + 4 B-frags -> 8 WMMAs.
// Double-buffered LDS: prefetch next panel to regs, compute, commit, 1 barrier.
template <bool OUT_F32>
__global__ __launch_bounds__(256) void gemm_bf16_wmma(const bf16* __restrict__ A,
                                                      const bf16* __restrict__ Bt,
                                                      void* __restrict__ C,
                                                      int M, int Nd, int Kd) {
    __shared__ __align__(16) bf16 sA[2][128][40];  // 80B stride: conflict-free b128
    __shared__ __align__(16) bf16 sB[2][128][40];
    const int tid  = threadIdx.x;
    const int wid  = tid >> 5, lane = tid & 31;
    const int m    = lane & 15, hh = lane >> 4;
    const int mi   = wid >> 1,  ni = wid & 1;
    const int row0 = blockIdx.y * 128;
    const int col0 = blockIdx.x * 128;
    // staging: 512 b128 chunks per 128x32 panel, 2 per thread per matrix
    const int r0c = tid >> 2,           off0 = (tid & 3) * 8;
    const int r1c = (tid + 256) >> 2,   off1 = ((tid + 256) & 3) * 8;
    const bf16* gA0 = A  + (size_t)(row0 + r0c) * Kd + off0;
    const bf16* gA1 = A  + (size_t)(row0 + r1c) * Kd + off1;
    const bf16* gB0 = Bt + (size_t)(col0 + r0c) * Kd + off0;
    const bf16* gB1 = Bt + (size_t)(col0 + r1c) * Kd + off1;

    *(u32x4*)&sA[0][r0c][off0] = *(const u32x4*)gA0;    // prologue: panel 0
    *(u32x4*)&sA[0][r1c][off1] = *(const u32x4*)gA1;
    *(u32x4*)&sB[0][r0c][off0] = *(const u32x4*)gB0;
    *(u32x4*)&sB[0][r1c][off1] = *(const u32x4*)gB1;

    v8f acc[2][4] = {};
    int cur = 0;
    for (int kk = 0; kk < Kd; kk += 32, cur ^= 1) {
        __syncthreads();
        const bool nxt = (kk + 32) < Kd;
        u32x4 pA0, pA1, pB0, pB1;                       // prefetch next panel
        if (nxt) {
            pA0 = *(const u32x4*)(gA0 + kk + 32);
            pA1 = *(const u32x4*)(gA1 + kk + 32);
            pB0 = *(const u32x4*)(gB0 + kk + 32);
            pB1 = *(const u32x4*)(gB1 + kk + 32);
        }
        Frag a0 = load_frag(&sA[cur][32 * mi + m][0], hh);
        Frag a1 = load_frag(&sA[cur][32 * mi + 16 + m][0], hh);
        Frag b[4];
#pragma unroll
        for (int j = 0; j < 4; ++j)
            b[j] = load_frag(&sB[cur][64 * ni + 16 * j + m][0], hh);
#pragma unroll
        for (int j = 0; j < 4; ++j) {
            acc[0][j] = wmma_bf16(a0, b[j], acc[0][j]);
            acc[1][j] = wmma_bf16(a1, b[j], acc[1][j]);
        }
        if (nxt) {                                      // commit to other buffer
            *(u32x4*)&sA[cur ^ 1][r0c][off0] = pA0;
            *(u32x4*)&sA[cur ^ 1][r1c][off1] = pA1;
            *(u32x4*)&sB[cur ^ 1][r0c][off0] = pB0;
            *(u32x4*)&sB[cur ^ 1][r1c][off1] = pB1;
        }
    }
#pragma unroll
    for (int i = 0; i < 2; ++i)
#pragma unroll
        for (int r = 0; r < 8; ++r) {
            size_t rr = (size_t)(row0 + 32 * mi + 16 * i + r + 8 * hh) * Nd;
#pragma unroll
            for (int j = 0; j < 4; ++j) {
                int cc = col0 + 64 * ni + 16 * j + m;
                if constexpr (OUT_F32) ((float*)C)[rr + cc] = acc[i][j][r];
                else                   ((bf16*)C)[rr + cc]  = (bf16)acc[i][j][r];
            }
        }
}

// ----------------------------------------------------------------- RoPE -----
// QKV bf16 [4096][2304] -> Qh [b][h][n][64] (x0.125), Kh [b][h][n][64],
// Vt [b][h][64][n]  (V transposed so P@V B-frags are contiguous loads).
__global__ __launch_bounds__(256) void rope_split(
        const bf16* __restrict__ QKV,
        const float* __restrict__ cos_t, const float* __restrict__ sin_t,
        const float* __restrict__ cos_h, const float* __restrict__ sin_h,
        const float* __restrict__ cos_w, const float* __restrict__ sin_w,
        const int* __restrict__ t_idx, const int* __restrict__ h_idx,
        const int* __restrict__ w_idx,
        bf16* __restrict__ Qh, bf16* __restrict__ Kh, bf16* __restrict__ Vt) {
    const int row = blockIdx.x;              // 0..4095 = b*2048 + n
    const int n = row & 2047, b = row >> 11;
    const int ti = t_idx[n], hi = h_idx[n], wi = w_idx[n];
    const bf16* qr = QKV + (size_t)row * 2304;
#pragma unroll
    for (int j = 0; j < 3; ++j) {
        int d = threadIdx.x + j * 256;       // 0..767
        float c, s, sgn;
        int part;
        if (d < 192) {                       // temporal section, half = 96
            c = cos_t[ti * 192 + d]; s = sin_t[ti * 192 + d];
            if (d < 96) { part = d + 96; sgn = -1.f; } else { part = d - 96; sgn = 1.f; }
        } else if (d < 480) {                // height section, half = 144
            int dd = d - 192;
            c = cos_h[hi * 288 + dd]; s = sin_h[hi * 288 + dd];
            if (dd < 144) { part = d + 144; sgn = -1.f; } else { part = d - 144; sgn = 1.f; }
        } else {                             // width section, half = 144
            int dd = d - 480;
            c = cos_w[wi * 288 + dd]; s = sin_w[wi * 288 + dd];
            if (dd < 144) { part = d + 144; sgn = -1.f; } else { part = d - 144; sgn = 1.f; }
        }
        float q  = (float)qr[d],        qp = (float)qr[part];
        float k  = (float)qr[768 + d],  kp = (float)qr[768 + part];
        float v  = (float)qr[1536 + d];
        float oq = (q * c + sgn * qp * s) * 0.125f;   // fold 1/sqrt(64)
        float ok =  k * c + sgn * kp * s;
        int hd = d >> 6, dd2 = d & 63;
        size_t base = ((size_t)(b * 12 + hd) * 2048 + n) * 64 + dd2;
        Qh[base] = (bf16)oq;
        Kh[base] = (bf16)ok;
        Vt[((size_t)(b * 12 + hd) * 64 + dd2) * 2048 + n] = (bf16)v;
    }
}

// ----------------------------------------------------- flash attention ------
// One WG = one (b,h) x 64-query block. 4 waves, each owns 16 query rows.
// S tile 16x64 via 8 WMMAs, online softmax (row max via step-major batched
// bpermute butterflies, row sum via WMMA vs all-ones frag), O += P@V (8 WMMAs).
__global__ __launch_bounds__(128) void flash_attn(const bf16* __restrict__ Qh,
                                                  const bf16* __restrict__ Kh,
                                                  const bf16* __restrict__ Vt,
                                                  bf16* __restrict__ Ob) {
    __shared__ __align__(16) bf16 sP[4][16][80];   // wave-private P slabs, padded
    const int bh = blockIdx.y;                     // 0..23
    const int b = bh / 12, h = bh % 12;
    const int wid = threadIdx.x >> 5, lane = threadIdx.x & 31;
    const int m = lane & 15, hh = lane >> 4;
    const int q0 = blockIdx.x * 64 + wid * 16;
    const bf16* Q = Qh + (size_t)bh * 2048 * 64;
    const bf16* K = Kh + (size_t)bh * 2048 * 64;
    const bf16* V = Vt + (size_t)bh * 64 * 2048;

    Frag qa[2];                                    // persistent Q fragments
#pragma unroll
    for (int kf = 0; kf < 2; ++kf)
        qa[kf] = load_frag(Q + (size_t)(q0 + m) * 64 + kf * 32, hh);

    Frag ones;                                     // B-frag of 1.0: P@ones = rowsum
#pragma unroll
    for (int i = 0; i < 16; ++i) ones.v[i] = (bf16)1.0f;

    float rmax[8];
    v8f o[4] = {};
    v8f osum = {};                                 // running softmax denominator
#pragma unroll
    for (int r = 0; r < 8; ++r) rmax[r] = -1e30f;

    for (int kb = 0; kb < 2048; kb += 64) {
        // ---- S = Q @ K^T for a 16x64 tile
        v8f S[4];
#pragma unroll
        for (int nt = 0; nt < 4; ++nt) {
            const bf16* kp = K + (size_t)(kb + nt * 16 + m) * 64;
            Frag b0 = load_frag(kp, hh);           // d 0..31
            Frag b1 = load_frag(kp + 32, hh);      // d 32..63
            v8f s = {};
            s = wmma_bf16(qa[0], b0, s);
            s = wmma_bf16(qa[1], b1, s);
            S[nt] = s;
        }
        // ---- row max: per-lane max over tiles, then step-major 16-lane butterfly
        float t[8];
#pragma unroll
        for (int r = 0; r < 8; ++r)
            t[r] = fmaxf(fmaxf(S[0][r], S[1][r]), fmaxf(S[2][r], S[3][r]));
#pragma unroll
        for (int st = 1; st <= 8; st <<= 1) {      // 8 bpermutes in flight / step
#pragma unroll
            for (int r = 0; r < 8; ++r)
                t[r] = fmaxf(t[r], __shfl_xor(t[r], st, 32));
        }
        // ---- online rescale + exponentials
#pragma unroll
        for (int r = 0; r < 8; ++r) {
            float nm  = fmaxf(rmax[r], t[r]);
            float fac = __expf(rmax[r] - nm);
            rmax[r] = nm;
#pragma unroll
            for (int nt = 0; nt < 4; ++nt)
                S[nt][r] = __expf(S[nt][r] - nm);
            osum[r] *= fac;
#pragma unroll
            for (int dt = 0; dt < 4; ++dt) o[dt][r] *= fac;
        }
        // ---- P (C layout) -> LDS -> A-fragment layout
#pragma unroll
        for (int nt = 0; nt < 4; ++nt)
#pragma unroll
            for (int r = 0; r < 8; ++r)
                sP[wid][r + 8 * hh][nt * 16 + m] = (bf16)S[nt][r];
        __syncthreads();
        Frag pa[2];
#pragma unroll
        for (int kf = 0; kf < 2; ++kf)
            pa[kf] = load_frag(&sP[wid][m][kf * 32], hh);
        // ---- rowsum via matrix unit: osum += P @ 1
        osum = wmma_bf16(pa[0], ones, osum);
        osum = wmma_bf16(pa[1], ones, osum);
        // ---- O += P @ V  (Vt rows = d, cols = keys: contiguous B-frag loads)
#pragma unroll
        for (int dt = 0; dt < 4; ++dt) {
            const bf16* vp = V + (size_t)(dt * 16 + m) * 2048 + kb;
            Frag v0 = load_frag(vp, hh);           // keys kb..kb+31
            Frag v1 = load_frag(vp + 32, hh);      // keys kb+32..kb+63
            o[dt] = wmma_bf16(pa[0], v0, o[dt]);
            o[dt] = wmma_bf16(pa[1], v1, o[dt]);
        }
        __syncthreads();                           // WAR on sP before next block
    }
    // ---- normalize, store to (B*N, C) bf16 for the output projection
#pragma unroll
    for (int r = 0; r < 8; ++r) {
        float inv = 1.0f / osum[r];
        size_t grow = (size_t)(b * 2048 + q0 + r + 8 * hh) * 768;
#pragma unroll
        for (int dt = 0; dt < 4; ++dt)
            Ob[grow + h * 64 + dt * 16 + m] = (bf16)(o[dt][r] * inv);
    }
}

// ---------------------------------------------------------------- launch ----
extern "C" void kernel_launch(void* const* d_in, const int* in_sizes, int n_in,
                              void* d_out, int out_size, void* d_ws, size_t ws_size,
                              hipStream_t stream) {
    (void)in_sizes; (void)n_in; (void)out_size; (void)ws_size;
    const float* x     = (const float*)d_in[0];
    const float* Wqkv  = (const float*)d_in[1];
    const float* Wout  = (const float*)d_in[2];
    const float* cos_t = (const float*)d_in[3];
    const float* sin_t = (const float*)d_in[4];
    const float* cos_h = (const float*)d_in[5];
    const float* sin_h = (const float*)d_in[6];
    const float* cos_w = (const float*)d_in[7];
    const float* sin_w = (const float*)d_in[8];
    const int*   t_idx = (const int*)d_in[9];
    const int*   h_idx = (const int*)d_in[10];
    const int*   w_idx = (const int*)d_in[11];

    char*  ws  = (char*)d_ws;
    size_t off = 0;
    auto carve = [&](size_t bytes) {
        void* p = ws + off;
        off += (bytes + 255) & ~(size_t)255;
        return p;
    };
    bf16* Xb     = (bf16*)carve(4096ull * 768 * 2);    // x in bf16
    bf16* Wqkv_t = (bf16*)carve(2304ull * 768 * 2);    // Wqkv^T bf16
    bf16* Wout_t = (bf16*)carve(768ull * 768 * 2);     // Wout^T bf16
    bf16* QKV    = (bf16*)carve(4096ull * 2304 * 2);   // qkv bf16
    bf16* Qhb    = (bf16*)carve(2ull * 12 * 2048 * 64 * 2);
    bf16* Khb    = (bf16*)carve(2ull * 12 * 2048 * 64 * 2);
    bf16* Vtb    = (bf16*)carve(2ull * 12 * 64 * 2048 * 2);
    bf16* Ob     = Xb;   // Xb dead after GEMM1 -> reuse for attention output

    convert_f32_bf16<<<(4096 * 768) / 256, 256, 0, stream>>>(x, Xb, 4096 * 768);
    transpose_to_bf16<<<dim3(2304 / 32, 768 / 32), dim3(32, 8), 0, stream>>>(Wqkv, Wqkv_t, 768, 2304);
    transpose_to_bf16<<<dim3(768 / 32, 768 / 32), dim3(32, 8), 0, stream>>>(Wout, Wout_t, 768, 768);

    gemm_bf16_wmma<false><<<dim3(2304 / 128, 4096 / 128), 256, 0, stream>>>(
        Xb, Wqkv_t, (void*)QKV, 4096, 2304, 768);

    rope_split<<<4096, 256, 0, stream>>>(QKV, cos_t, sin_t, cos_h, sin_h, cos_w, sin_w,
                                         t_idx, h_idx, w_idx, Qhb, Khb, Vtb);

    flash_attn<<<dim3(32, 24), 128, 0, stream>>>(Qhb, Khb, Vtb, Ob);

    gemm_bf16_wmma<true><<<dim3(768 / 128, 4096 / 128), 256, 0, stream>>>(
        Ob, Wout_t, d_out, 4096, 768, 768);
}
